// VectorQuantizer_78632261255735
// MI455X (gfx1250) — matrix operand order repaired and checked by reading, not compiled, not run on hardware
//
#include <hip/hip_runtime.h>

// ---------------- problem constants ----------------
#define DDIM 512                 // embedding dim
#define KCB  8192                // codebook entries
#define NROW 32768               // B*S = 8*4096 rows
#define NQE  (NROW * DDIM)       // quantized element count (16,777,216)

#define CONTRIB 0.05f
#define CCOST   0.25f

typedef __attribute__((ext_vector_type(16))) __bf16          v16bf;
typedef __attribute__((ext_vector_type(8)))  float           v8f;
typedef __attribute__((ext_vector_type(4)))  unsigned int    u32x4;
typedef __attribute__((ext_vector_type(4)))  unsigned short  u16x4;

union FragBF { v16bf bf; u32x4 q[2]; };

__device__ __forceinline__ unsigned short bf16_rne(float f) {
  union { float f; unsigned u; } v; v.f = f;
  unsigned u = v.u;
  u += 0x7FFFu + ((u >> 16) & 1u);     // round-to-nearest-even to bf16
  return (unsigned short)(u >> 16);
}
__device__ __forceinline__ float bf16_f32(unsigned short h) {
  union { unsigned u; float f; } v; v.u = ((unsigned)h) << 16;
  return v.f;
}

// ---------------------------------------------------------------------------
// Kernel 1: inputs fp32 -> bf16 hi/lo split (A operand)
// ---------------------------------------------------------------------------
__global__ void __launch_bounds__(256) vq_convert_inputs(
    const float* __restrict__ x,
    unsigned short* __restrict__ hi,
    unsigned short* __restrict__ lo) {
  const int n4 = NQE / 4;
  for (int i = blockIdx.x * blockDim.x + threadIdx.x; i < n4;
       i += gridDim.x * blockDim.x) {
    float4 v = ((const float4*)x)[i];
    u16x4 h, l;
    h.x = bf16_rne(v.x); l.x = bf16_rne(v.x - bf16_f32(h.x));
    h.y = bf16_rne(v.y); l.y = bf16_rne(v.y - bf16_f32(h.y));
    h.z = bf16_rne(v.z); l.z = bf16_rne(v.z - bf16_f32(h.z));
    h.w = bf16_rne(v.w); l.w = bf16_rne(v.w - bf16_f32(h.w));
    ((u16x4*)hi)[i] = h;
    ((u16x4*)lo)[i] = l;
  }
}

// ---------------------------------------------------------------------------
// Kernel 2: embedding fp32 -> bf16 hi/lo split (B operand) + row norms
// one wave per codebook row (32 lanes x 16 elems = 512)
// ---------------------------------------------------------------------------
__global__ void __launch_bounds__(256) vq_convert_embed(
    const float* __restrict__ e,
    unsigned short* __restrict__ hi,
    unsigned short* __restrict__ lo,
    float* __restrict__ enorm) {
  const int wave = threadIdx.x >> 5;
  const int lane = threadIdx.x & 31;
  const int row  = blockIdx.x * 8 + wave;
  const float* er = e + (size_t)row * DDIM + lane * 16;
  unsigned short* hr = hi + (size_t)row * DDIM + lane * 16;
  unsigned short* lr = lo + (size_t)row * DDIM + lane * 16;
  float nsum = 0.f;
#pragma unroll
  for (int c = 0; c < 4; ++c) {
    float4 v = ((const float4*)er)[c];
    nsum += v.x * v.x + v.y * v.y + v.z * v.z + v.w * v.w;
    u16x4 h, l;
    h.x = bf16_rne(v.x); l.x = bf16_rne(v.x - bf16_f32(h.x));
    h.y = bf16_rne(v.y); l.y = bf16_rne(v.y - bf16_f32(h.y));
    h.z = bf16_rne(v.z); l.z = bf16_rne(v.z - bf16_f32(h.z));
    h.w = bf16_rne(v.w); l.w = bf16_rne(v.w - bf16_f32(h.w));
    ((u16x4*)hr)[c] = h;
    ((u16x4*)lr)[c] = l;
  }
#pragma unroll
  for (int m = 16; m >= 1; m >>= 1) nsum += __shfl_xor(nsum, m, 32);
  if (lane == 0) enorm[row] = nsum;
}

// ---------------------------------------------------------------------------
// Kernel 3: main VQ kernel.
// Block = 256 threads (8 waves), owns 64 rows. Waves stride over the 512
// column tiles (16 cols each). Per column tile: accumulate dot over D=512
// with split-bf16 WMMA (3 per 16x16x32 step), then fold eNorm - 2*dot into
// a per-lane running argmin. Cross-lane + cross-wave argmin reduction, then
// gather + straight-through mix + loss partial.
// ---------------------------------------------------------------------------
__global__ void __launch_bounds__(256) vq_main(
    const float* __restrict__ x, const float* __restrict__ emb,
    const unsigned short* __restrict__ Ahi, const unsigned short* __restrict__ Alo,
    const unsigned short* __restrict__ Bhi, const unsigned short* __restrict__ Blo,
    const float* __restrict__ eNorm, float* __restrict__ out,
    float* __restrict__ lossPartial) {
  __shared__ float sVal[8][64];
  __shared__ int   sIdx[8][64];
  __shared__ int   sRow[64];
  __shared__ float sLoss[8];

  const int tid     = threadIdx.x;
  const int wave    = tid >> 5;
  const int lane    = tid & 31;
  const int laneCol = lane & 15;
  const int half    = lane >> 4;          // 0: lanes 0-15, 1: lanes 16-31
  const int ibase   = blockIdx.x * 64;

  float minv[4][8];
  int   mini[4][8];
#pragma unroll
  for (int mt = 0; mt < 4; ++mt)
#pragma unroll
    for (int r = 0; r < 8; ++r) { minv[mt][r] = 3.4e38f; mini[mt][r] = 0; }

  // Per-lane A row base (row = ibase + mt*16 + laneCol).
  const unsigned short* aHiBase = Ahi + (size_t)(ibase + laneCol) * DDIM;
  const unsigned short* aLoBase = Alo + (size_t)(ibase + laneCol) * DDIM;

  const v8f vzero = {};

  for (int ct = wave; ct < KCB / 16; ct += 8) {
    const int jcol = ct * 16 + laneCol;   // this lane's codebook column
    const unsigned short* bHiRow = Bhi + (size_t)jcol * DDIM + half * 16;
    const unsigned short* bLoRow = Blo + (size_t)jcol * DDIM + half * 16;

    v8f acc[4];
#pragma unroll
    for (int mt = 0; mt < 4; ++mt) acc[mt] = vzero;

    for (int d0 = 0; d0 < DDIM; d0 += 32) {
      // B fragment: 16 contiguous bf16 (32B) per lane; lanes 16-31 take K+16.
      FragBF bh, bl;
      bh.q[0] = *(const u32x4*)(bHiRow + d0);
      bh.q[1] = *(const u32x4*)(bHiRow + d0 + 8);
      bl.q[0] = *(const u32x4*)(bLoRow + d0);
      bl.q[1] = *(const u32x4*)(bLoRow + d0 + 8);
#pragma unroll
      for (int mt = 0; mt < 4; ++mt) {
        // A fragment: elems 0-15 contiguous at d0+half*16.
        const unsigned short* ah = aHiBase + (size_t)mt * 16 * DDIM + d0 + half * 16;
        const unsigned short* al = aLoBase + (size_t)mt * 16 * DDIM + d0 + half * 16;
        FragBF fah, fal;
        fah.q[0] = *(const u32x4*)(ah);
        fah.q[1] = *(const u32x4*)(ah + 8);
        fal.q[0] = *(const u32x4*)(al);
        fal.q[1] = *(const u32x4*)(al + 8);
        acc[mt] = __builtin_amdgcn_wmma_f32_16x16x32_bf16(
            false, fah.bf, false, bh.bf, (short)0, acc[mt], false, false);
        acc[mt] = __builtin_amdgcn_wmma_f32_16x16x32_bf16(
            false, fah.bf, false, bl.bf, (short)0, acc[mt], false, false);
        acc[mt] = __builtin_amdgcn_wmma_f32_16x16x32_bf16(
            false, fal.bf, false, bh.bf, (short)0, acc[mt], false, false);
      }
    }
    // dist' = ||e_j||^2 - 2*dot  (||x||^2 constant per row, irrelevant to argmin)
    const float en = eNorm[jcol];
#pragma unroll
    for (int mt = 0; mt < 4; ++mt)
#pragma unroll
      for (int r = 0; r < 8; ++r) {
        float dv = en - 2.0f * acc[mt][r];
        if (dv < minv[mt][r]) { minv[mt][r] = dv; mini[mt][r] = jcol; }
      }
  }

  // C layout: reg r, lane L -> row = mt*16 + r + (L>=16)*8, col = L%16.
  // Reduce (val,idx) across each 16-lane half; ties -> smaller index (argmin
  // first-occurrence semantics).
#pragma unroll
  for (int mt = 0; mt < 4; ++mt)
#pragma unroll
    for (int r = 0; r < 8; ++r) {
      float v = minv[mt][r];
      int   i = mini[mt][r];
#pragma unroll
      for (int m = 8; m >= 1; m >>= 1) {
        float v2 = __shfl_xor(v, m, 32);
        int   i2 = __shfl_xor(i, m, 32);
        if (v2 < v || (v2 == v && i2 < i)) { v = v2; i = i2; }
      }
      if (lane == 0)  { sVal[wave][mt * 16 + r]     = v; sIdx[wave][mt * 16 + r]     = i; }
      if (lane == 16) { sVal[wave][mt * 16 + r + 8] = v; sIdx[wave][mt * 16 + r + 8] = i; }
    }
  __syncthreads();

  if (tid < 64) {
    float bv = sVal[0][tid];
    int   bi = sIdx[0][tid];
#pragma unroll
    for (int w = 1; w < 8; ++w) {
      float v2 = sVal[w][tid];
      int   i2 = sIdx[w][tid];
      if (v2 < bv || (v2 == bv && i2 < bi)) { bv = v2; bi = i2; }
    }
    sRow[tid] = bi;
    out[NQE + ibase + tid] = (float)bi;   // encoding_indices as float
  }
  __syncthreads();

  // Gather + straight-through mix + loss.
  // out = x*(1-CONTRIB) + (q - x)*CONTRIB = 0.90*x + 0.05*q
  float lsum = 0.f;
#pragma unroll
  for (int rr = 0; rr < 8; ++rr) {
    const int lrow = wave * 8 + rr;
    const int grow = ibase + lrow;
    const int ei   = sRow[lrow];
    const float4* xr = (const float4*)(x + (size_t)grow * DDIM);
    const float4* er = (const float4*)(emb + (size_t)ei * DDIM);
    float4* orow = (float4*)(out + (size_t)grow * DDIM);
    for (int c = lane; c < DDIM / 4; c += 32) {
      float4 xv = xr[c];
      float4 ev = er[c];
      float4 ov;
      ov.x = (1.f - 2.f * CONTRIB) * xv.x + CONTRIB * ev.x;
      ov.y = (1.f - 2.f * CONTRIB) * xv.y + CONTRIB * ev.y;
      ov.z = (1.f - 2.f * CONTRIB) * xv.z + CONTRIB * ev.z;
      ov.w = (1.f - 2.f * CONTRIB) * xv.w + CONTRIB * ev.w;
      float dx = ev.x - xv.x; lsum += dx * dx;
      dx = ev.y - xv.y; lsum += dx * dx;
      dx = ev.z - xv.z; lsum += dx * dx;
      dx = ev.w - xv.w; lsum += dx * dx;
      orow[c] = ov;
    }
  }
#pragma unroll
  for (int m = 16; m >= 1; m >>= 1) lsum += __shfl_xor(lsum, m, 32);
  if (lane == 0) sLoss[wave] = lsum;
  __syncthreads();
  if (tid == 0) {
    float t = 0.f;
#pragma unroll
    for (int w = 0; w < 8; ++w) t += sLoss[w];
    lossPartial[blockIdx.x] = t;
  }
}

// ---------------------------------------------------------------------------
// Kernel 4: deterministic reduction of 512 block partials -> loss scalar.
// loss = q_latent + CCOST*e_latent = (1+CCOST) * mean((q-x)^2)
// ---------------------------------------------------------------------------
__global__ void __launch_bounds__(256) vq_finalize(
    const float* __restrict__ partials, float* __restrict__ out) {
  __shared__ float s[256];
  s[threadIdx.x] = partials[threadIdx.x] + partials[threadIdx.x + 256];
  __syncthreads();
  for (int st = 128; st > 0; st >>= 1) {
    if (threadIdx.x < st) s[threadIdx.x] += s[threadIdx.x + st];
    __syncthreads();
  }
  if (threadIdx.x == 0)
    out[NQE + NROW] = (1.0f + CCOST) * s[0] / (float)NQE;
}

// ---------------------------------------------------------------------------
extern "C" void kernel_launch(void* const* d_in, const int* in_sizes, int n_in,
                              void* d_out, int out_size, void* d_ws, size_t ws_size,
                              hipStream_t stream) {
  const float* x   = (const float*)d_in[0];   // [8,4096,512] fp32
  const float* emb = (const float*)d_in[1];   // [8192,512]  fp32
  float* out = (float*)d_out;
  char* ws = (char*)d_ws;

  // workspace layout (bytes)
  unsigned short* Ahi = (unsigned short*)(ws);
  unsigned short* Alo = (unsigned short*)(ws + (size_t)NQE * 2);
  unsigned short* Bhi = (unsigned short*)(ws + (size_t)NQE * 4);
  unsigned short* Blo = (unsigned short*)(ws + (size_t)NQE * 4 + (size_t)KCB * DDIM * 2);
  float* eNorm        = (float*)(ws + (size_t)NQE * 4 + (size_t)KCB * DDIM * 4);
  float* lossPartial  = eNorm + KCB;

  vq_convert_inputs<<<NQE / 4 / 256, 256, 0, stream>>>(x, Ahi, Alo);
  vq_convert_embed<<<KCB / 8, 256, 0, stream>>>(emb, Bhi, Blo, eNorm);
  vq_main<<<NROW / 64, 256, 0, stream>>>(x, emb, Ahi, Alo, Bhi, Blo, eNorm,
                                         out, lossPartial);
  vq_finalize<<<1, 256, 0, stream>>>(lossPartial, out);
}
